// Mamba_61125974557132
// MI455X (gfx1250) — matrix operand order repaired and checked by reading, not compiled, not run on hardware
//
#include <hip/hip_runtime.h>
#include <math.h>

// ---- problem constants (match reference) ----
constexpr int cH   = 1024;   // hidden
constexpr int cDI  = 2048;   // inner dim
constexpr int cN   = 16;     // SSM state
constexpr int cR   = 64;     // dt rank
constexpr int cL   = 2048;   // sequence length
constexpr int cK   = 4;      // conv width
constexpr int cXD  = cR + 2 * cN;  // 96, x_dbl row width
constexpr int cNCH = 16;     // scan chunks
constexpr int cCHL = cL / cNCH;  // 128

// ---- CDNA5 WMMA types ----
typedef __attribute__((ext_vector_type(16))) __bf16 v16bf;
typedef __attribute__((ext_vector_type(8)))  float  v8f;
typedef __attribute__((ext_vector_type(2)))  float  v2f;

union FragBF { v16bf v; uint4 q[2]; };
union Acc8   { v8f v; float f[8]; };

__device__ __forceinline__ unsigned short f2bf(float f) {
  unsigned int u = __float_as_uint(f);
  u += 0x7FFFu + ((u >> 16) & 1u);            // round-to-nearest-even
  return (unsigned short)(u >> 16);
}
__device__ __forceinline__ float sigm(float x) { return 1.0f / (1.0f + __expf(-x)); }

// ---------------- elementwise converts ----------------
__global__ void k_f32_to_bf16(const float* __restrict__ src,
                              unsigned short* __restrict__ dst, int n) {
  int i = blockIdx.x * blockDim.x + threadIdx.x;
  if (i < n) dst[i] = f2bf(src[i]);
}

__global__ void k_aneg(const float* __restrict__ alog, float* __restrict__ aneg, int n) {
  int i = blockIdx.x * blockDim.x + threadIdx.x;
  if (i < n) aneg[i] = -__expf(alog[i]);
}

// ---------------- register-blocked bf16 WMMA GEMM ----------------
// C[M,N] = A[M,K] * W[N,K]^T ; each wave computes (MBT*16) x (NBT*16) outputs.
// MODE 0: C[row*ldc+col] = acc (+bias)
// MODE 1: in_proj split epilogue: col <  cDI -> C  (xc, raw + bias)
//                                 col >= cDI -> C2 (silu(acc+bias)), i.e. silu(res)
template <int MBT, int NBT, int MODE>
__global__ void k_gemm_bf16_blk(const unsigned short* __restrict__ A, int lda,
                                const unsigned short* __restrict__ W, int ldw,
                                float* __restrict__ C, int ldc,
                                float* __restrict__ C2,
                                const float* __restrict__ bias,
                                int MtB, int NtB, int Kdim) {
  const int wave = blockIdx.x * (blockDim.x >> 5) + (threadIdx.x >> 5);
  const int lane = threadIdx.x & 31;
  if (wave >= MtB * NtB) return;               // wave-uniform: EXEC stays all-ones
  const int nb = (wave % NtB) * (NBT * 16);
  const int mb = (wave / NtB) * (MBT * 16);
  const int m  = lane & 15;
  const int hi = lane >> 4;                    // half-wave select
  const unsigned short* Ar[MBT];
  const unsigned short* Wr[NBT];
#pragma unroll
  for (int i = 0; i < MBT; ++i)
    Ar[i] = A + (size_t)(mb + i * 16 + m) * lda + hi * 8;    // K {0..7,16..23}/{8..15,24..31}
#pragma unroll
  for (int j = 0; j < NBT; ++j)
    Wr[j] = W + (size_t)(nb + j * 16 + m) * ldw + hi * 8;    // K {0..7,16..23}/{8..15,24..31}

  Acc8 acc[MBT][NBT] = {};
  for (int kb = 0; kb < Kdim; kb += 32) {
    FragBF a[MBT], b[NBT];
#pragma unroll
    for (int i = 0; i < MBT; ++i) {
      a[i].q[0] = *(const uint4*)(Ar[i] + kb);
      a[i].q[1] = *(const uint4*)(Ar[i] + kb + 16);
      __builtin_prefetch(Ar[i] + kb + 128, 0, 1);            // global_prefetch_b8
    }
#pragma unroll
    for (int j = 0; j < NBT; ++j) {
      b[j].q[0] = *(const uint4*)(Wr[j] + kb);
      b[j].q[1] = *(const uint4*)(Wr[j] + kb + 16);
      __builtin_prefetch(Wr[j] + kb + 128, 0, 1);
    }
#pragma unroll
    for (int i = 0; i < MBT; ++i)
#pragma unroll
      for (int j = 0; j < NBT; ++j)
        acc[i][j].v = __builtin_amdgcn_wmma_f32_16x16x32_bf16(
            false, a[i].v, false, b[j].v, (short)0, acc[i][j].v, false, false);
  }

#pragma unroll
  for (int i = 0; i < MBT; ++i) {
#pragma unroll
    for (int j = 0; j < NBT; ++j) {
      const int col = nb + j * 16 + m;
      const float badd = bias ? bias[col] : 0.0f;
      const int row0 = mb + i * 16 + hi * 8;
#pragma unroll
      for (int r = 0; r < 8; ++r) {
        const float v = acc[i][j].f[r] + badd;
        if (MODE == 0) {
          C[(size_t)(row0 + r) * ldc + col] = v;
        } else {
          if (col < cDI) C[(size_t)(row0 + r) * cDI + col] = v;
          else           C2[(size_t)(row0 + r) * cDI + (col - cDI)] = v * sigm(v);
        }
      }
    }
  }
}

// ---------------- fp32 WMMA dt_proj + softplus ----------------
// Delta[L,DI] = softplus( Xdbl[:, 0:64] * Wdt[DI,64]^T + bdt )
__global__ void k_dtproj_softplus(const float* __restrict__ Xdbl,
                                  const float* __restrict__ Wdt,
                                  const float* __restrict__ bdt,
                                  float* __restrict__ Delta) {
  const int wave = blockIdx.x * (blockDim.x >> 5) + (threadIdx.x >> 5);
  const int lane = threadIdx.x & 31;
  const int Nt = cDI / 16, Mt = cL / 16;
  if (wave >= Mt * Nt) return;
  const int nt = wave % Nt, mt = wave / Nt;
  const int m  = lane & 15;
  const int hi = lane >> 4;
  const float* Arow = Xdbl + (size_t)(mt * 16 + m) * cXD;   // delta_in row (stride 96)
  const float* Wrow = Wdt  + (size_t)(nt * 16 + m) * cR;
  Acc8 acc{};
#pragma unroll
  for (int kb = 0; kb < cR; kb += 4) {        // v_wmma_f32_16x16x4_f32, 16 steps
    v2f a = *(const v2f*)(Arow + kb + hi * 2);
    v2f b = *(const v2f*)(Wrow + kb + hi * 2);
    acc.v = __builtin_amdgcn_wmma_f32_16x16x4_f32(
        false, a, false, b, (short)0, acc.v, false, false);
  }
  const float bb = bdt[nt * 16 + m];
  float* Crow = Delta + (size_t)(mt * 16 + hi * 8) * cDI + nt * 16 + m;
#pragma unroll
  for (int r = 0; r < 8; ++r) {
    float x = acc.f[r] + bb;
    Crow[(size_t)r * cDI] = (x > 20.0f) ? x : __logf(1.0f + __expf(x));
  }
}

// ---------------- causal depthwise conv1d + SiLU, LDS-staged ----------------
// Block covers CLT l-values x CDT channels; xc tile (+ causal halo) staged in LDS,
// each global element read exactly once; 4 taps served from LDS (ds_load).
constexpr int CLT = 64, CDT = 64;
__global__ void k_conv_silu_lds(const float* __restrict__ XC,  // [L, DI]
                                const float* __restrict__ CW,  // [DI, 1, K]
                                const float* __restrict__ CB,  // [DI]
                                float* __restrict__ Xs,        // [L, DI] fp32
                                unsigned short* __restrict__ XsBf) {
  __shared__ float tile[(CLT + cK - 1) * CDT];
  const int l0 = blockIdx.x * CLT;
  const int d0 = blockIdx.y * CDT;
  // cooperative load: rows l0-(K-1) .. l0+CLT-1, cols d0..d0+CDT-1 (zero-pad l<0)
  for (int i = threadIdx.x; i < (CLT + cK - 1) * CDT; i += blockDim.x) {
    const int r = i / CDT, dc = i % CDT;
    const int l = l0 - (cK - 1) + r;
    tile[i] = (l >= 0) ? XC[(size_t)l * cDI + d0 + dc] : 0.0f;
  }
  __syncthreads();
  const int dc = threadIdx.x & (CDT - 1);
  const int d  = d0 + dc;
  const int lt = threadIdx.x >> 6;             // 4 groups, each covers CLT/4 rows
  float w[cK];
#pragma unroll
  for (int k = 0; k < cK; ++k) w[k] = CW[d * cK + k];
  const float b = CB[d];
  for (int j = 0; j < CLT / 4; ++j) {
    const int lr = lt * (CLT / 4) + j;         // local output row; taps = rows lr..lr+3
    float acc = b;
#pragma unroll
    for (int k = 0; k < cK; ++k) acc += tile[(lr + k) * CDT + dc] * w[k];
    const float s = acc * sigm(acc);
    const size_t o = (size_t)(l0 + lr) * cDI + d;
    Xs[o]   = s;
    XsBf[o] = f2bf(s);
  }
}

// ---------------- selective scan, chunked linear recurrence ----------------
__global__ void k_scan1(const float* __restrict__ Delta, const float* __restrict__ Xdbl,
                        const float* __restrict__ Xs, const float* __restrict__ Aneg,
                        float* __restrict__ Pc, float* __restrict__ Sc) {
  const int idx = blockIdx.x * blockDim.x + threadIdx.x;  // c*DI + d
  const int d = idx & (cDI - 1);
  const int c = idx >> 11;
  float An[cN], P[cN], S[cN];
#pragma unroll
  for (int n = 0; n < cN; ++n) { An[n] = Aneg[d * cN + n]; P[n] = 1.0f; S[n] = 0.0f; }
  const int l0 = c * cCHL;
  for (int l = l0; l < l0 + cCHL; ++l) {
    const float dl = Delta[(size_t)l * cDI + d];
    const float xv = Xs[(size_t)l * cDI + d];
    const float* Bl = Xdbl + (size_t)l * cXD + cR;   // B columns 64..79
    const float dx = dl * xv;
#pragma unroll
    for (int n = 0; n < cN; ++n) {
      const float dA = __expf(dl * An[n]);
      P[n] *= dA;
      S[n] = S[n] * dA + dx * Bl[n];
    }
  }
  float* pp = Pc + (size_t)idx * cN;
  float* ss = Sc + (size_t)idx * cN;
#pragma unroll
  for (int n = 0; n < cN; ++n) { pp[n] = P[n]; ss[n] = S[n]; }
}

__global__ void k_scan2(const float* __restrict__ Delta, const float* __restrict__ Xdbl,
                        const float* __restrict__ Xs, const float* __restrict__ Aneg,
                        const float* __restrict__ Pc, const float* __restrict__ Sc,
                        const float* __restrict__ Dv, const float* __restrict__ Sres,
                        unsigned short* __restrict__ Ybf) {
  const int idx = blockIdx.x * blockDim.x + threadIdx.x;  // c*DI + d
  const int d = idx & (cDI - 1);
  const int c = idx >> 11;
  float An[cN], h[cN];
#pragma unroll
  for (int n = 0; n < cN; ++n) { An[n] = Aneg[d * cN + n]; h[n] = 0.0f; }
  for (int cc = 0; cc < c; ++cc) {            // incoming state: h = P*h + S per chunk
    const float* pp = Pc + ((size_t)cc * cDI + d) * cN;
    const float* ss = Sc + ((size_t)cc * cDI + d) * cN;
#pragma unroll
    for (int n = 0; n < cN; ++n) h[n] = h[n] * pp[n] + ss[n];
  }
  const float dd = Dv[d];
  const int l0 = c * cCHL;
  for (int l = l0; l < l0 + cCHL; ++l) {
    const float dl = Delta[(size_t)l * cDI + d];
    const float xv = Xs[(size_t)l * cDI + d];
    const float* Bl = Xdbl + (size_t)l * cXD + cR;        // 64..79
    const float* Cl = Xdbl + (size_t)l * cXD + cR + cN;   // 80..95
    const float dx = dl * xv;
    float y = 0.0f;
#pragma unroll
    for (int n = 0; n < cN; ++n) {
      const float dA = __expf(dl * An[n]);
      h[n] = h[n] * dA + dx * Bl[n];
      y += h[n] * Cl[n];
    }
    y = (y + xv * dd) * Sres[(size_t)l * cDI + d];
    Ybf[(size_t)l * cDI + d] = f2bf(y);
  }
}

extern "C" void kernel_launch(void* const* d_in, const int* in_sizes, int n_in,
                              void* d_out, int out_size, void* d_ws, size_t ws_size,
                              hipStream_t stream) {
  const float* x      = (const float*)d_in[0];   // [1, L, H]
  const float* w_in   = (const float*)d_in[1];   // [2*DI, H]
  const float* b_in   = (const float*)d_in[2];   // [2*DI]
  const float* conv_w = (const float*)d_in[3];   // [DI, 1, K]
  const float* conv_b = (const float*)d_in[4];   // [DI]
  const float* w_xp   = (const float*)d_in[5];   // [R+2N, DI]
  const float* w_dt   = (const float*)d_in[6];   // [DI, R]
  const float* b_dt   = (const float*)d_in[7];   // [DI]
  const float* a_log  = (const float*)d_in[8];   // [DI, N]
  const float* Dvec   = (const float*)d_in[9];   // [DI]
  const float* w_out  = (const float*)d_in[10];  // [H, DI]
  float* out = (float*)d_out;                    // [1, L, H]

  // workspace carve-out
  char* wsp = (char*)d_ws;
  auto alloc = [&](size_t bytes) -> char* {
    char* p = wsp; wsp += (bytes + 255) & ~(size_t)255; return p;
  };
  unsigned short* xbf   = (unsigned short*)alloc((size_t)cL * cH * 2);
  unsigned short* w1bf  = (unsigned short*)alloc((size_t)2 * cDI * cH * 2);
  unsigned short* xpbf  = (unsigned short*)alloc((size_t)cXD * cDI * 2);
  unsigned short* wobf  = (unsigned short*)alloc((size_t)cH * cDI * 2);
  float*          xc    = (float*)alloc((size_t)cL * cDI * 4);   // pre-conv half
  float*          sres  = (float*)alloc((size_t)cL * cDI * 4);   // silu(res), fused epilogue
  float*          xs    = (float*)alloc((size_t)cL * cDI * 4);
  unsigned short* xsbf  = (unsigned short*)alloc((size_t)cL * cDI * 2);
  float*          xdbl  = (float*)alloc((size_t)cL * cXD * 4);
  float*          delta = (float*)alloc((size_t)cL * cDI * 4);
  float*          aneg  = (float*)alloc((size_t)cDI * cN * 4);
  float*          Pc    = (float*)alloc((size_t)cNCH * cDI * cN * 4);
  float*          Sc    = (float*)alloc((size_t)cNCH * cDI * cN * 4);
  unsigned short* ybf   = (unsigned short*)alloc((size_t)cL * cDI * 2);

  const int TB = 256;
  auto cgrid = [](int n, int tb) { return (n + tb - 1) / tb; };

  // 1) bf16 packing of activations + weights
  k_f32_to_bf16<<<cgrid(cL * cH, TB), TB, 0, stream>>>(x, xbf, cL * cH);
  k_f32_to_bf16<<<cgrid(2 * cDI * cH, TB), TB, 0, stream>>>(w_in, w1bf, 2 * cDI * cH);
  k_f32_to_bf16<<<cgrid(cXD * cDI, TB), TB, 0, stream>>>(w_xp, xpbf, cXD * cDI);
  k_f32_to_bf16<<<cgrid(cH * cDI, TB), TB, 0, stream>>>(w_out, wobf, cH * cDI);
  k_aneg<<<cgrid(cDI * cN, TB), TB, 0, stream>>>(a_log, aneg, cDI * cN);

  // 2) in_proj, 64x64 per wave, split epilogue: xc (raw) / sres = silu(res)
  {
    const int MtB = cL / 64, NtB = (2 * cDI) / 64;    // 32 x 64 wave-blocks
    k_gemm_bf16_blk<4, 4, 1><<<(MtB * NtB) / 8, TB, 0, stream>>>(
        xbf, cH, w1bf, cH, xc, cDI, sres, b_in, MtB, NtB, cH);
  }
  // 3) causal conv + SiLU via LDS tile (each xc element read once from global)
  {
    dim3 g(cL / CLT, cDI / CDT);
    k_conv_silu_lds<<<g, TB, 0, stream>>>(xc, conv_w, conv_b, xs, xsbf);
  }
  // 4) x_proj: xdbl[L, 96] = xs @ w_xp^T (one A-frag reused over all 6 N tiles)
  {
    const int MtB = cL / 16, NtB = 1;                 // 128 x 1 wave-blocks (16x96 each)
    k_gemm_bf16_blk<1, 6, 0><<<(MtB * NtB) / 8, TB, 0, stream>>>(
        xsbf, cDI, xpbf, cDI, xdbl, cXD, nullptr, nullptr, MtB, NtB, cDI);
  }
  // 5) dt_proj + softplus: delta[L, DI]  (fp32 WMMA 16x16x4, precision-critical)
  {
    const int Mt = cL / 16, Nt = cDI / 16;
    k_dtproj_softplus<<<(Mt * Nt) / 8, TB, 0, stream>>>(xdbl, w_dt, b_dt, delta);
  }
  // 6) chunked selective scan (state in registers, no [L,DI,N] materialization)
  k_scan1<<<cgrid(cNCH * cDI, TB), TB, 0, stream>>>(delta, xdbl, xs, aneg, Pc, Sc);
  k_scan2<<<cgrid(cNCH * cDI, TB), TB, 0, stream>>>(delta, xdbl, xs, aneg, Pc, Sc,
                                                    Dvec, sres, ybf);
  // 7) out_proj: out[L, H] = y @ w_out^T, 64x64 per wave
  {
    const int MtB = cL / 64, NtB = cH / 64;           // 32 x 16 wave-blocks
    k_gemm_bf16_blk<4, 4, 0><<<(MtB * NtB) / 8, TB, 0, stream>>>(
        ybf, cDI, wobf, cDI, out, cH, nullptr, nullptr, MtB, NtB, cDI);
  }
}
